// AttentionFusion_60601988546962
// MI455X (gfx1250) — compile-verified, matrix operand order checked by reading, not run on hardware
//
#include <hip/hip_runtime.h>
#include <hip/hip_bf16.h>

typedef __bf16 bf16_t;
typedef __attribute__((ext_vector_type(16))) __bf16 v16bf;
typedef __attribute__((ext_vector_type(8)))  __bf16 v8bf;
typedef __attribute__((ext_vector_type(4)))  __bf16 v4bf;
typedef __attribute__((ext_vector_type(8)))  float  v8f;
typedef unsigned int u32x4 __attribute__((ext_vector_type(4)));
typedef unsigned int u32x8 __attribute__((ext_vector_type(8)));

#define BM 128
#define BN 128
#define BKT 32
#define LDSK 40   // BKT + 8 pad: row stride 80B -> every 16B fragment read stays 16B-aligned

// Load one 16x32 bf16 WMMA operand fragment from an LDS tile stored row-major
// (16 rows x LDSK). Per ISA 16-bit A/B layout: lanes 0-15 hold K 0..7 & 16..23,
// lanes 16-31 hold K 8..15 & 24..31 of row (lane & 15). Two ds_load_b128 each.
__device__ __forceinline__ v16bf load_frag_nk(const bf16_t* tileBase, int lane) {
    const int laneRow = lane & 15;
    const int hsel    = (lane >> 4) & 1;
    const bf16_t* r = tileBase + laneRow * LDSK;
    v8bf lo = *(const v8bf*)(r + hsel * 8);
    v8bf hi = *(const v8bf*)(r + 16 + hsel * 8);
    return __builtin_shufflevector(lo, hi, 0,1,2,3,4,5,6,7,8,9,10,11,12,13,14,15);
}

template<typename T> __device__ __forceinline__ T cvt_out(float v);
template<> __device__ __forceinline__ float  cvt_out<float >(float v) { return v; }
template<> __device__ __forceinline__ bf16_t cvt_out<bf16_t>(float v) { return (bf16_t)v; }

// Truncate a generic pointer to a shared-memory object down to its 32-bit LDS
// byte offset (generic LDS addresses are {aperture_hi32, lds_offset32}).
__device__ __forceinline__ unsigned lds_off(const void* p) {
    return (unsigned)(unsigned long long)(uintptr_t)p;
}

// C[M,N] = scale * A[M,K] x B + bias.  A row-major bf16.
// A tile staged by the Tensor Data Mover (wave 0 issues a 2D D#; TDM applies
// the 16B row padding itself via pad_interval/pad_amount).
// B_IS_NK=true : B memory is [N,K] row-major; staged with per-lane
//                global_load_async_to_lds_b128 (ASYNCcnt).
// B_IS_NK=false: B memory is [K,N] row-major; transposed while staging via LDS.
// Block: 256 thr = 8 waves; tile 128x128x32; wave tile 64x32 = 4x2 WMMA.
template<bool B_IS_NK, typename OutT, bool BIAS>
__global__ __launch_bounds__(256)
void gemm_bf16_wmma(const bf16_t* __restrict__ A, const bf16_t* __restrict__ B,
                    OutT* __restrict__ C, const float* __restrict__ bias,
                    int M, int N, int K, int lda, int ldb, int ldc, float scale,
                    long sAb, long sBb, long sCb)
{
    __shared__ alignas(16) bf16_t As[BM][LDSK];
    __shared__ alignas(16) bf16_t Bs[BN][LDSK];   // always N-major: Bs[n][k]

    const int bz = blockIdx.z;
    A += (long)bz * sAb;  B += (long)bz * sBb;  C += (long)bz * sCb;

    const int tid   = threadIdx.x;
    const int lane  = tid & 31;
    const int wave  = tid >> 5;
    const int waveM = wave >> 2;   // 0..1 -> 64 rows
    const int waveN = wave & 3;    // 0..3 -> 32 cols
    const int blockRow = blockIdx.x * BM;
    const int blockCol = blockIdx.y * BN;

    const unsigned ldsA = lds_off(&As[0][0]);

    v8f acc[4][2] = {};

    for (int k0 = 0; k0 < K; k0 += BKT) {
        // ---- stage A tile (BM x BKT) via TDM: one 2D descriptor from wave 0.
        if (wave == 0) {
            const unsigned long long ga =
                (unsigned long long)(uintptr_t)(A + (long)blockRow * lda + k0);
            u32x4 g0;
            g0[0] = 1u;                                   // count=1 user descriptor
            g0[1] = ldsA;                                 // lds_addr (bytes)
            g0[2] = (unsigned)(ga & 0xffffffffull);       // global_addr[31:0]
            g0[3] = (unsigned)((ga >> 32) & 0x1ffffffull) // global_addr[56:32]
                    | (2u << 30);                         // type=2 ("image")
            u32x8 g1;
            // data_size=2B; pad_enable, pad_interval=16 DW (64B), pad_amount=4 DW (16B)
            g1[0] = (1u << 16) | (1u << 20) | (3u << 22) | (3u << 25);
            const unsigned td0 = (unsigned)K;             // tensor_dim0
            const unsigned td1 = (unsigned)M;             // tensor_dim1
            g1[1] = (td0 & 0xffffu) << 16;                // bits 63:48
            g1[2] = (td0 >> 16) | ((td1 & 0xffffu) << 16);// bits 79:64, 95:80
            g1[3] = (td1 >> 16) | ((unsigned)BKT << 16);  // bits111:96, tile_dim0
            g1[4] = (unsigned)BM;                         // tile_dim1 (tile_dim2=0)
            g1[5] = (unsigned)lda;                        // tensor_dim0_stride[31:0]
            g1[6] = 0u;                                   // stride[47:32], dim1_stride lo
            g1[7] = 0u;
            asm volatile("tensor_load_to_lds %0, %1" :: "s"(g0), "s"(g1) : "memory");
            __builtin_amdgcn_s_wait_tensorcnt(0);
        }
        // ---- stage B tile into Bs[n][k]
        if constexpr (B_IS_NK) {
            const int r = tid >> 1;
            const int c = (tid & 1) << 4;
            const bf16_t* g = B + (long)(blockCol + r) * ldb + k0 + c;
            const unsigned     ldsB = lds_off(&Bs[r][c]);
            const unsigned long long ga = (unsigned long long)(uintptr_t)g;
            // IOFFSET is added to both the LDS and global addresses.
            asm volatile("global_load_async_to_lds_b128 %0, %1, off"
                         :: "v"(ldsB), "v"(ga) : "memory");
            asm volatile("global_load_async_to_lds_b128 %0, %1, off offset:16"
                         :: "v"(ldsB), "v"(ga) : "memory");
            if (k0 + BKT < K) __builtin_prefetch(g + BKT, 0, 1);  // global_prefetch_b8
            asm volatile("s_wait_asynccnt 0x0" ::: "memory");
        } else {
            const int kk = tid >> 3;        // 0..31
            const int n0 = (tid & 7) << 4;  // 0..112
            const bf16_t* g = B + (long)(k0 + kk) * ldb + blockCol + n0;
            v8bf b0 = *(const v8bf*)(g);
            v8bf b1 = *(const v8bf*)(g + 8);
            #pragma unroll
            for (int i = 0; i < 8; ++i) Bs[n0 + i][kk]     = b0[i];
            #pragma unroll
            for (int i = 0; i < 8; ++i) Bs[n0 + 8 + i][kk] = b1[i];
            if (k0 + BKT < K) __builtin_prefetch(g + (long)BKT * ldb, 0, 1);
        }
        __syncthreads();

        v16bf afrag[4], bfrag[2];
        #pragma unroll
        for (int i = 0; i < 4; ++i)
            afrag[i] = load_frag_nk(&As[waveM * 64 + i * 16][0], lane);
        #pragma unroll
        for (int j = 0; j < 2; ++j)
            bfrag[j] = load_frag_nk(&Bs[waveN * 32 + j * 16][0], lane);
        #pragma unroll
        for (int i = 0; i < 4; ++i)
            #pragma unroll
            for (int j = 0; j < 2; ++j)
                acc[i][j] = __builtin_amdgcn_wmma_f32_16x16x32_bf16(
                    false, afrag[i], false, bfrag[j],
                    (short)0, acc[i][j], false, false);
        __syncthreads();
    }

    // Epilogue per C/D layout: lanes 0-15 -> rows 0..7, lanes 16-31 -> rows 8..15
    const int laneRow = lane & 15;
    const int rsel    = (lane >> 4) & 1;
    #pragma unroll
    for (int j = 0; j < 2; ++j) {
        const int col = blockCol + waveN * 32 + j * 16 + laneRow;
        const float bv = BIAS ? bias[col] : 0.0f;
        #pragma unroll
        for (int i = 0; i < 4; ++i) {
            #pragma unroll
            for (int r = 0; r < 8; ++r) {
                const int row = blockRow + waveM * 64 + i * 16 + rsel * 8 + r;
                C[(long)row * ldc + col] = cvt_out<OutT>(acc[i][j][r] * scale + bv);
            }
        }
    }
}

__global__ __launch_bounds__(256)
void f32_to_bf16_kernel(const float* __restrict__ in, bf16_t* __restrict__ out, long n) {
    const long i = ((long)blockIdx.x * blockDim.x + threadIdx.x) * 4;
    if (i + 3 < n) {
        float4 v = *(const float4*)(in + i);
        v4bf o;
        o[0] = (bf16_t)v.x; o[1] = (bf16_t)v.y; o[2] = (bf16_t)v.z; o[3] = (bf16_t)v.w;
        *(v4bf*)(out + i) = o;
    }
}

// In-place row softmax over bf16 scores; one 256-thread block per row of N=2048.
__global__ __launch_bounds__(256)
void softmax_rows_kernel(bf16_t* __restrict__ S, int N) {
    __shared__ float red[256];
    const long row = blockIdx.x;
    bf16_t* p = S + row * (long)N;
    const int t = threadIdx.x;

    v8bf x = *(const v8bf*)(p + t * 8);
    float v[8];
    float m = -1e30f;
    #pragma unroll
    for (int i = 0; i < 8; ++i) { v[i] = (float)x[i]; m = fmaxf(m, v[i]); }
    red[t] = m; __syncthreads();
    for (int s = 128; s > 0; s >>= 1) { if (t < s) red[t] = fmaxf(red[t], red[t + s]); __syncthreads(); }
    m = red[0]; __syncthreads();

    float sum = 0.0f;
    #pragma unroll
    for (int i = 0; i < 8; ++i) { v[i] = __expf(v[i] - m); sum += v[i]; }
    red[t] = sum; __syncthreads();
    for (int s = 128; s > 0; s >>= 1) { if (t < s) red[t] += red[t + s]; __syncthreads(); }
    const float inv = 1.0f / red[0];

    v8bf o;
    #pragma unroll
    for (int i = 0; i < 8; ++i) o[i] = (bf16_t)(v[i] * inv);
    *(v8bf*)(p + t * 8) = o;
}

// lambda = sigmoid(ho_row . W_lambda + b); ct = lambda*ho + (1-lambda)*attended -> bf16
__global__ __launch_bounds__(256)
void gate_ct_kernel(const float* __restrict__ ho, const float* __restrict__ att,
                    const float* __restrict__ Wl, const float* __restrict__ bl,
                    bf16_t* __restrict__ ct, int H) {
    __shared__ float red[256];
    const long row = blockIdx.x;
    const float* h = ho + row * (long)H;
    const float* a = att + row * (long)H;
    const int t = threadIdx.x;

    float4 hv = *(const float4*)(h + t * 4);
    float4 wv = *(const float4*)(Wl + t * 4);
    red[t] = hv.x * wv.x + hv.y * wv.y + hv.z * wv.z + hv.w * wv.w;
    __syncthreads();
    for (int s = 128; s > 0; s >>= 1) { if (t < s) red[t] += red[t + s]; __syncthreads(); }
    const float lam = 1.0f / (1.0f + __expf(-(red[0] + bl[0])));

    float4 av = *(const float4*)(a + t * 4);
    v4bf o;
    o[0] = (bf16_t)(lam * hv.x + (1.0f - lam) * av.x);
    o[1] = (bf16_t)(lam * hv.y + (1.0f - lam) * av.y);
    o[2] = (bf16_t)(lam * hv.z + (1.0f - lam) * av.z);
    o[3] = (bf16_t)(lam * hv.w + (1.0f - lam) * av.w);
    *(v4bf*)(ct + row * (long)H + t * 4) = o;
}

extern "C" void kernel_launch(void* const* d_in, const int* in_sizes, int n_in,
                              void* d_out, int out_size, void* d_ws, size_t ws_size,
                              hipStream_t stream) {
    (void)in_sizes; (void)n_in; (void)out_size; (void)ws_size;
    const float* ho = (const float*)d_in[0];
    const float* hi = (const float*)d_in[1];
    const float* Wl = (const float*)d_in[2];
    const float* bl = (const float*)d_in[3];
    const float* Wp = (const float*)d_in[4];
    const float* bp = (const float*)d_in[5];
    float* out = (float*)d_out;

    const int  Bb = 4, S = 2048, H = 1024, O = 512;
    const long SH = (long)S * H;   // 2,097,152
    const long SS = (long)S * S;   // 4,194,304
    const size_t MB = 1024u * 1024u;

    char* ws = (char*)d_ws;
    bf16_t* ho_b = (bf16_t*)(ws);             // 16 MB
    bf16_t* hi_b = (bf16_t*)(ws + 16 * MB);   // 16 MB
    bf16_t* scP  = (bf16_t*)(ws + 32 * MB);   // 32 MB (scores, softmax'd in place -> P)
    float*  att  = (float*) (ws + 64 * MB);   // 32 MB
    bf16_t* ct   = (bf16_t*)(ws + 96 * MB);   // 16 MB
    bf16_t* wp_b = (bf16_t*)(ws + 112 * MB);  //  1 MB

    const long nconv = (long)Bb * SH;                 // 8,388,608
    f32_to_bf16_kernel<<<dim3((unsigned)(nconv / 1024)), 256, 0, stream>>>(ho, ho_b, nconv);
    f32_to_bf16_kernel<<<dim3((unsigned)(nconv / 1024)), 256, 0, stream>>>(hi, hi_b, nconv);
    const long nw = (long)H * O;                      // 524,288
    f32_to_bf16_kernel<<<dim3((unsigned)(nw / 1024)), 256, 0, stream>>>(Wp, wp_b, nw);

    // scores = (ho . hi^T) / 32 -> bf16   [per batch: 2048x2048x1024]
    gemm_bf16_wmma<true, bf16_t, false><<<dim3(S / BM, S / BN, Bb), 256, 0, stream>>>(
        ho_b, hi_b, scP, nullptr, S, S, H, H, H, S, 0.03125f, SH, SH, SS);

    // row softmax in place -> P
    softmax_rows_kernel<<<dim3(Bb * S), 256, 0, stream>>>(scP, S);

    // attended = P . hi -> f32            [per batch: 2048x1024x2048]
    gemm_bf16_wmma<false, float, false><<<dim3(S / BM, H / BN, Bb), 256, 0, stream>>>(
        scP, hi_b, att, nullptr, S, H, S, S, H, H, 1.0f, SS, SH, SH);

    // ct = lambda*ho + (1-lambda)*attended -> bf16
    gate_ct_kernel<<<dim3(Bb * S), 256, 0, stream>>>(ho, att, Wl, bl, ct, H);

    // out = ct . W_proj + b_proj -> f32   [8192x512x1024]
    gemm_bf16_wmma<false, float, true><<<dim3((Bb * S) / BM, O / BN, 1), 256, 0, stream>>>(
        ct, wp_b, out, bp, Bb * S, O, H, H, O, O, 1.0f, 0, 0, 0);
}
